// Attn_head_1726576854471
// MI455X (gfx1250) — compile-verified, matrix operand order checked by reading, not run on hardware
//
#include <hip/hip_runtime.h>
#include <hip/hip_bf16.h>

#define N    8192
#define CIN  128
#define COUT 64

typedef __attribute__((ext_vector_type(16))) __bf16        v16bf;
typedef __attribute__((ext_vector_type(8)))  float         v8f;
typedef __attribute__((ext_vector_type(8)))  unsigned int  v8u;

__device__ __forceinline__ unsigned short f2bf(float x) {
    unsigned int u = __float_as_uint(x);
    u += 0x7fffu + ((u >> 16) & 1u);   // round-to-nearest-even
    return (unsigned short)(u >> 16);
}

// K1: seq_fts[o][n] = sum_i w1[o][i] * x[i][n]
__global__ void k_seqfts(const float* __restrict__ x, const float* __restrict__ w1,
                         float* __restrict__ seq) {
    int o = blockIdx.x >> 5;                               // 64
    int n = ((blockIdx.x & 31) << 8) + threadIdx.x;        // 8192
    const float* wr = w1 + o * CIN;
    float acc = 0.f;
    #pragma unroll 8
    for (int i = 0; i < CIN; ++i) acc += wr[i] * x[(size_t)i * N + n];
    seq[(size_t)o * N + n] = acc;
}

// K2: f[n] = sum_o w2[o] * seq_fts[o][n]
__global__ void k_f(const float* __restrict__ seq, const float* __restrict__ w2,
                    float* __restrict__ f) {
    int n = blockIdx.x * 256 + threadIdx.x;
    float acc = 0.f;
    #pragma unroll
    for (int o = 0; o < COUT; ++o) acc += w2[o] * seq[(size_t)o * N + n];
    f[n] = acc;
}

// K3: per-column partial softmax stats over an i-chunk (online max/sum),
// 4 columns per thread via float4 loads (4 independent exp chains).
__global__ void k_stats(const float* __restrict__ bias, const float* __restrict__ f,
                        float* __restrict__ pm, float* __restrict__ ps) {
    int j0 = (blockIdx.x * 256 + threadIdx.x) * 4;   // gridDim.x = 8
    int ic = blockIdx.y;                             // 32 chunks of 256 rows
    const float4 fj4 = *(const float4*)(f + j0);
    float fj[4] = {fj4.x, fj4.y, fj4.z, fj4.w};
    float m[4], s[4];
    #pragma unroll
    for (int c = 0; c < 4; ++c) { m[c] = -3.4e38f; s[c] = 0.f; }
    int i0 = ic * 256;
    for (int i = i0; i < i0 + 256; ++i) {
        float fi = f[i];
        const float4 b4 = *(const float4*)(bias + (size_t)i * N + j0);
        float bv[4] = {b4.x, b4.y, b4.z, b4.w};
        #pragma unroll
        for (int c = 0; c < 4; ++c) {
            float l = fi + fj[c];
            l = l > 0.f ? l : 0.01f * l;              // leaky_relu(0.01)
            l += bv[c];
            if (l > m[c]) { s[c] = s[c] * __expf(m[c] - l) + 1.f; m[c] = l; }
            else          { s[c] += __expf(l - m[c]); }
        }
    }
    *(float4*)(pm + (size_t)ic * N + j0) = make_float4(m[0], m[1], m[2], m[3]);
    *(float4*)(ps + (size_t)ic * N + j0) = make_float4(s[0], s[1], s[2], s[3]);
}

// K4: combine partial stats -> m_j, 1/Z_j
__global__ void k_red(const float* __restrict__ pm, const float* __restrict__ ps,
                      float* __restrict__ mcol, float* __restrict__ invZ) {
    int j = blockIdx.x * 256 + threadIdx.x;
    float m = -3.4e38f;
    for (int c = 0; c < 32; ++c) m = fmaxf(m, pm[(size_t)c * N + j]);
    float z = 0.f;
    for (int c = 0; c < 32; ++c) z += ps[(size_t)c * N + j] * __expf(pm[(size_t)c * N + j] - m);
    mcol[j] = m;
    invZ[j] = 1.f / z;
}

// K5: build B-operand fragments: V[j][o] = seq_fts[o][j] / Z_j, bf16, pre-swizzled
// into WMMA 16-bit B layout (lane = col + 16*khalf, VGPR v = k-pair).
__global__ void k_vfrag(const float* __restrict__ seq, const float* __restrict__ invZ,
                        unsigned int* __restrict__ vfrag) {
    int t     = blockIdx.x * 256 + threadIdx.x;  // 32768 = 256 jchunks * 4 otiles * 32 lanes
    int jc    = t >> 7;
    int r     = t & 127;
    int otile = r >> 5;
    int lane  = r & 31;
    int h     = lane >> 4;
    int o     = otile * 16 + (lane & 15);
    unsigned int* dst = vfrag + ((size_t)(jc * 4 + otile) * 32 + lane) * 8;
    #pragma unroll
    for (int v = 0; v < 8; ++v) {
        int kb = (v < 4 ? 2 * v : 16 + 2 * (v - 4)) + 8 * h;
        int j0 = jc * 32 + kb;
        unsigned short lo = f2bf(seq[(size_t)o * N + j0]     * invZ[j0]);
        unsigned short hi = f2bf(seq[(size_t)o * N + j0 + 1] * invZ[j0 + 1]);
        dst[v] = (unsigned int)lo | ((unsigned int)hi << 16);
    }
}

// K6: fused exp + GEMM.  Block = 128 i-rows x 64 o, 8 waves; split-K=4 over j.
// A[i,j] = exp(leaky(f[i]+f[j]) + bias[i,j] - m_j) built in LDS in A-fragment order.
// B-fragments are software-pipelined one iteration ahead so their global loads
// are covered by the exp/LDS production phase + barrier.
__global__ __launch_bounds__(256) void k_gemm(const float* __restrict__ bias,
                                              const float* __restrict__ f,
                                              const float* __restrict__ mcol,
                                              const unsigned int* __restrict__ vfrag,
                                              float* __restrict__ part) {
    __shared__ float fi[128];
    __shared__ float fj[32];
    __shared__ float mj[32];
    __shared__ unsigned int asw[8 * 32 * 8];   // [wavetile][lane][vgpr] bf16-pairs

    int tid  = threadIdx.x;
    int wave = tid >> 5, lane = tid & 31;
    int i0    = blockIdx.x * 128;
    int jbase = blockIdx.y * 2048;
    int jend  = jbase + 2048;

    if (tid < 128) fi[tid] = f[i0 + tid];

    v8f acc[4];
    #pragma unroll
    for (int t = 0; t < 4; ++t)
        #pragma unroll
        for (int k = 0; k < 8; ++k) acc[t][k] = 0.f;

    // preload first B-fragments
    v8u bcur[4];
    {
        const unsigned int* vb = vfrag + (size_t)(jbase >> 5) * 4 * 256 + lane * 8;
        #pragma unroll
        for (int t = 0; t < 4; ++t) bcur[t] = *(const v8u*)(vb + (size_t)t * 256);
    }

    for (int jc = jbase; jc < jend; jc += 32) {
        __syncthreads();                                  // previous tile consumed
        if (tid < 32) { fj[tid] = f[jc + tid]; mj[tid] = mcol[jc + tid]; }
        __syncthreads();

        // produce bf16 A tile (128x32) into swizzled LDS
        #pragma unroll
        for (int q = 0; q < 4; ++q) {
            int slot = tid + 256 * q;
            int row  = slot >> 3;
            int c4   = (slot & 7) << 2;
            const float4 b4 = *(const float4*)(bias + (size_t)(i0 + row) * N + jc + c4);
            if (((tid & 7) == 0) && (jc + 32 < N))        // next-tile row prefetch (128B)
                __builtin_prefetch(bias + (size_t)(i0 + row) * N + jc + 32, 0, 0);
            float fir = fi[row];
            float bv[4] = {b4.x, b4.y, b4.z, b4.w};
            #pragma unroll
            for (int e = 0; e < 4; ++e) {
                int col = c4 + e;
                float l = fir + fj[col];
                l = l > 0.f ? l : 0.01f * l;
                l += bv[e];
                unsigned short ab = f2bf(__expf(l - mj[col]));
                int wt = row >> 4;
                int rr = row & 15;
                int hh = (col >> 3) & 1;
                int v  = ((col & 16) ? 4 : 0) + ((col & 7) >> 1);
                int lane2 = rr + (hh << 4);
                unsigned short* p = (unsigned short*)&asw[(wt * 32 + lane2) * 8 + v];
                p[col & 1] = ab;
            }
        }

        // prefetch next iteration's B-fragments (latency hidden by barrier + A produce)
        bool more = (jc + 32) < jend;
        v8u bnext[4];
        if (more) {
            const unsigned int* vb = vfrag + (size_t)((jc + 32) >> 5) * 4 * 256 + lane * 8;
            #pragma unroll
            for (int t = 0; t < 4; ++t) bnext[t] = *(const v8u*)(vb + (size_t)t * 256);
        }

        __syncthreads();

        // consume: one A-fragment per wave, 4 WMMAs against pipelined B-fragments
        v16bf a = __builtin_bit_cast(v16bf, *(const v8u*)&asw[(wave * 32 + lane) * 8]);
        #pragma unroll
        for (int t = 0; t < 4; ++t) {
            v16bf b = __builtin_bit_cast(v16bf, bcur[t]);
            acc[t] = __builtin_amdgcn_wmma_f32_16x16x32_bf16(
                         false, a, false, b, (short)0, acc[t], false, false);
        }
        if (more) {
            #pragma unroll
            for (int t = 0; t < 4; ++t) bcur[t] = bnext[t];
        }
    }

    // write split-K partials: C layout lane=(col, M-half), VGPR r = M row
    int h  = lane >> 4, cL = lane & 15;
    int ir = i0 + wave * 16 + 8 * h;
    float* pb = part + ((size_t)blockIdx.y * N + ir) * COUT;
    #pragma unroll
    for (int t = 0; t < 4; ++t) {
        int o = t * 16 + cL;
        #pragma unroll
        for (int r = 0; r < 8; ++r)
            pb[(size_t)r * COUT + o] = acc[t][r];
    }
}

// K7: reduce split-K partials + elu, output [C_OUT][N]
__global__ void k_final(const float* __restrict__ part, float* __restrict__ out) {
    int idx = blockIdx.x * 256 + threadIdx.x;   // 524288
    int o = idx >> 13, i = idx & (N - 1);
    float s = 0.f;
    #pragma unroll
    for (int sp = 0; sp < 4; ++sp) s += part[((size_t)sp * N + i) * COUT + o];
    out[(size_t)o * N + i] = s > 0.f ? s : __expf(s) - 1.f;
}

extern "C" void kernel_launch(void* const* d_in, const int* in_sizes, int n_in,
                              void* d_out, int out_size, void* d_ws, size_t ws_size,
                              hipStream_t stream) {
    const float* x    = (const float*)d_in[0];
    const float* bias = (const float*)d_in[1];
    const float* w1   = (const float*)d_in[2];
    const float* w2   = (const float*)d_in[3];
    float* out = (float*)d_out;

    char* ws = (char*)d_ws;
    size_t off = 0;
    float* seq  = (float*)(ws + off); off += (size_t)COUT * N * 4;   // 2 MB
    float* f    = (float*)(ws + off); off += (size_t)N * 4;
    float* pm   = (float*)(ws + off); off += (size_t)32 * N * 4;     // 1 MB
    float* ps   = (float*)(ws + off); off += (size_t)32 * N * 4;     // 1 MB
    float* mcol = (float*)(ws + off); off += (size_t)N * 4;
    float* invZ = (float*)(ws + off); off += (size_t)N * 4;
    unsigned int* vfrag = (unsigned int*)(ws + off); off += (size_t)256 * 4 * 32 * 8 * 4; // 1 MB
    float* part = (float*)(ws + off); off += (size_t)4 * N * COUT * 4;  // 8 MB

    k_seqfts<<<dim3(2048),   dim3(256), 0, stream>>>(x, w1, seq);
    k_f     <<<dim3(32),     dim3(256), 0, stream>>>(seq, w2, f);
    k_stats <<<dim3(8, 32),  dim3(256), 0, stream>>>(bias, f, pm, ps);
    k_red   <<<dim3(32),     dim3(256), 0, stream>>>(pm, ps, mcol, invZ);
    k_vfrag <<<dim3(128),    dim3(256), 0, stream>>>(seq, invZ, vfrag);
    k_gemm  <<<dim3(64, 4),  dim3(256), 0, stream>>>(bias, f, mcol, vfrag, part);
    k_final <<<dim3(2048),   dim3(256), 0, stream>>>(part, out);
}
